// Performer_11682311045350
// MI455X (gfx1250) — matrix operand order repaired and checked by reading, not compiled, not run on hardware
//
#include <hip/hip_runtime.h>
#include <math.h>

typedef _Float16 half_t;
typedef __attribute__((ext_vector_type(16))) _Float16 v16h;
typedef __attribute__((ext_vector_type(8)))  _Float16 v8h;
typedef __attribute__((ext_vector_type(8)))  float    v8f;

#define WMMA_F16(a, b, c) \
  __builtin_amdgcn_wmma_f32_16x16x32_f16(false, (a), false, (b), (short)0, (c), false, false)

#define BB 2
#define NN 8192
#define DD 512
#define LL 2
#define HH 8
#define DHD 64
#define MMF 256

// ---------------- WMMA fragment loaders (ISA 7.12.2 layouts) ----------------
// A fragment 16x32 f16: lane<16 -> row=lane, K={kb..kb+7, kb+16..kb+23} kb=0
//                       lane>=16 -> row=lane-16, kb=8
static __device__ __forceinline__ v16h load_a_frag(const half_t* base, int lda) {
  int lane = threadIdx.x & 31;
  int row  = lane & 15;
  int kb   = (lane & 16) ? 8 : 0;
  const half_t* p = base + (size_t)row * lda + kb;
  union { v16h v; v8h h[2]; } u;
  u.h[0] = *(const v8h*)(p);
  u.h[1] = *(const v8h*)(p + 16);
  return u.v;
}
// B fragment 32x16 f16: lane<16 -> col=lane, K={kb..kb+7, kb+16..kb+23} kb=0; lane>=16 -> col=lane-16, kb=8
static __device__ __forceinline__ v16h load_b_frag(const half_t* base, int ldb) {
  int lane = threadIdx.x & 31;
  int kb   = (lane & 16) ? 8 : 0;
  const half_t* p = base + (size_t)kb * ldb + (lane & 15);
  union { v16h v; v8h h[2]; } u;
  for (int s = 0; s < 16; ++s) u.v[s] = p[(size_t)(s < 8 ? s : s + 8) * ldb];
  return u.v;
}

static __device__ __forceinline__ float gelu_f(float x) {
  float x3 = x * x * x;
  return 0.5f * x * (1.0f + tanhf(0.7978845608f * (x + 0.044715f * x3)));
}
static __device__ __forceinline__ unsigned f2o(float f) {
  unsigned u = __float_as_uint(f);
  return (u & 0x80000000u) ? ~u : (u | 0x80000000u);
}
static __device__ __forceinline__ float o2f(unsigned u) {
  unsigned b = (u & 0x80000000u) ? (u & 0x7FFFFFFFu) : ~u;
  return __uint_as_float(b);
}

// ---------------- generic f16 WMMA GEMM, 128x64 block tile ----------------
// 8 waves; wave w owns rows w*16..w*16+15, all 64 cols: 1 A-frag x 4 B-frags
// = 4 WMMAs per barrier pair. C = act(A@B + bias); f16 store or f32 residual add.
__global__ void __launch_bounds__(256)
gemm16(const half_t* __restrict__ A, int lda,
       const half_t* __restrict__ B, int ldb,
       const float* __restrict__ bias,
       half_t* __restrict__ Cout, float* __restrict__ Cres, int ldc,
       int K, int act)
{
  __shared__ __align__(16) half_t As[128][40];
  __shared__ __align__(16) half_t Bs[32][72];
  int m0 = blockIdx.x * 128, n0 = blockIdx.y * 64;
  int t = threadIdx.x;
  int ar0 = t >> 2, ac0 = (t & 3) * 8;          // A chunk 0 (rows 0..63)
  int ar1 = ar0 + 64;                            // A chunk 1 (rows 64..127)
  int br = t >> 3, bc = (t & 7) * 8;
  int wave = t >> 5, lane = t & 31;
  v8f acc[4] = {{}, {}, {}, {}};
  for (int k0 = 0; k0 < K; k0 += 32) {
    *(v8h*)&As[ar0][ac0] = *(const v8h*)&A[(size_t)(m0 + ar0) * lda + k0 + ac0];
    *(v8h*)&As[ar1][ac0] = *(const v8h*)&A[(size_t)(m0 + ar1) * lda + k0 + ac0];
    *(v8h*)&Bs[br][bc]   = *(const v8h*)&B[(size_t)(k0 + br) * ldb + n0 + bc];
    if (k0 + 32 < K) {
      __builtin_prefetch(&A[(size_t)(m0 + ar0) * lda + k0 + 32 + ac0], 0, 3);
      __builtin_prefetch(&B[(size_t)(k0 + 32 + br) * ldb + n0 + bc], 0, 3);
    }
    __syncthreads();
    v16h a = load_a_frag(&As[wave * 16][0], 40);
    for (int j = 0; j < 4; ++j) {
      v16h b = load_b_frag(&Bs[0][j * 16], 72);
      acc[j] = WMMA_F16(a, b, acc[j]);
    }
    __syncthreads();
  }
  int hi = (lane & 16) ? 8 : 0, cl = lane & 15;
  for (int j = 0; j < 4; ++j) {
    for (int r = 0; r < 8; ++r) {
      int row = m0 + wave * 16 + r + hi;
      int col = n0 + j * 16 + cl;
      float v = acc[j][r] + (bias ? bias[col] : 0.0f);
      if (act) v = gelu_f(v);
      if (Cres) Cres[(size_t)row * ldc + col] += v;
      else      Cout[(size_t)row * ldc + col] = (half_t)v;
    }
  }
}

// ---------------- FAVOR+ feature kernel ----------------
// dd[16 rows n][256 m] = X(16x64) @ projT(64x256) via WMMA, then
// mode 0: global max -> stab (ordered-uint atomicMax), for K pass1
// mode 1: row max, qp = ratio*(exp(dd-diag-rowmax)+eps)   [n-major out]
// mode 2: kp = ratio*(exp(dd-diag-stab)+eps)              [m-major (transposed) out]
__global__ void __launch_bounds__(256)
feat_k(const half_t* __restrict__ X, const half_t* __restrict__ projT,
       const float* __restrict__ diag, unsigned* __restrict__ stab,
       half_t* __restrict__ out, int mode)
{
  __shared__ float dd_s[16][257];
  __shared__ float red_s[256];
  __shared__ float st_s[16];
  __shared__ float dg_s[16];
  int n0 = blockIdx.x * 16;
  int bh = blockIdx.y;
  int b = bh >> 3, h = bh & 7;
  const half_t* A = X + ((size_t)(b * NN + n0)) * DD + h * DHD;
  int wave = threadIdx.x >> 5, lane = threadIdx.x & 31;
  int m0 = wave * 32;
  v8f acc0 = {}, acc1 = {};
  for (int k0 = 0; k0 < DHD; k0 += 32) {
    v16h a  = load_a_frag(A + k0, DD);
    v16h b0 = load_b_frag(projT + (size_t)k0 * MMF + m0, MMF);
    v16h b1 = load_b_frag(projT + (size_t)k0 * MMF + m0 + 16, MMF);
    acc0 = WMMA_F16(a, b0, acc0);
    acc1 = WMMA_F16(a, b1, acc1);
  }
  int hi = (lane & 16) ? 8 : 0, cl = lane & 15;
  for (int r = 0; r < 8; ++r) {
    dd_s[r + hi][m0 + cl]      = acc0[r];
    dd_s[r + hi][m0 + 16 + cl] = acc1[r];
  }
  __syncthreads();

  int row = threadIdx.x >> 4, j = threadIdx.x & 15;
  if (mode == 0) {
    float mx = -3.4e38f;
    for (int c = 0; c < 16; ++c) mx = fmaxf(mx, dd_s[row][j * 16 + c]);
    red_s[threadIdx.x] = mx;
    __syncthreads();
    if (threadIdx.x == 0) {
      float m = red_s[0];
      for (int i = 1; i < 256; ++i) m = fmaxf(m, red_s[i]);
      atomicMax(&stab[bh], f2o(m));
    }
  } else if (mode == 1) {
    float mx = -3.4e38f;
    for (int c = 0; c < 16; ++c) mx = fmaxf(mx, dd_s[row][j * 16 + c]);
    red_s[row * 16 + j] = mx;
    __syncthreads();
    if (j == 0) {
      float m = red_s[row * 16];
      for (int jj = 1; jj < 16; ++jj) m = fmaxf(m, red_s[row * 16 + jj]);
      st_s[row] = m;
    }
    __syncthreads();
    float dg = diag[(size_t)bh * NN + n0 + row];
    float st = st_s[row];
    size_t obase = ((size_t)bh * NN + n0 + row) * MMF;
    for (int c = 0; c < 16; ++c) {
      float v = 0.0625f * (expf(dd_s[row][j * 16 + c] - dg - st) + 1e-4f);
      out[obase + j * 16 + c] = (half_t)v;
    }
  } else {
    if (threadIdx.x < 16) dg_s[threadIdx.x] = diag[(size_t)bh * NN + n0 + threadIdx.x];
    __syncthreads();
    float st = o2f(stab[bh]);
    int m = threadIdx.x;
    size_t obase = ((size_t)bh * MMF + m) * NN + n0;
    for (int r = 0; r < 16; ++r) {
      float v = 0.0625f * (expf(dd_s[r][m] - dg_s[r] - st) + 1e-4f);
      out[obase + r] = (half_t)v;
    }
  }
}

// ---------------- ctx = kp^T @ v (split-K over n, f32 atomics) ----------------
// 128x64 tile: wave w owns rows w*16, 4 WMMAs per k-step
__global__ void __launch_bounds__(256)
ctx_k(const half_t* __restrict__ kpT, const half_t* __restrict__ v16,
      float* __restrict__ ctx)
{
  __shared__ __align__(16) half_t As[128][40];
  __shared__ __align__(16) half_t Bs[32][72];
  int m0 = blockIdx.x * 128;
  int nbase = blockIdx.y * 512;
  int bh = blockIdx.z, b = bh >> 3, h = bh & 7;
  const half_t* A  = kpT + ((size_t)bh * MMF + m0) * NN + nbase;
  const half_t* Bg = v16 + ((size_t)(b * NN + nbase)) * DD + h * DHD;
  int t = threadIdx.x;
  int ar0 = t >> 2, ac0 = (t & 3) * 8;
  int ar1 = ar0 + 64;
  int br = t >> 3, bc = (t & 7) * 8;
  int wave = t >> 5, lane = t & 31;
  v8f acc[4] = {{}, {}, {}, {}};
  for (int k0 = 0; k0 < 512; k0 += 32) {
    *(v8h*)&As[ar0][ac0] = *(const v8h*)&A[(size_t)ar0 * NN + k0 + ac0];
    *(v8h*)&As[ar1][ac0] = *(const v8h*)&A[(size_t)ar1 * NN + k0 + ac0];
    *(v8h*)&Bs[br][bc]   = *(const v8h*)&Bg[(size_t)(k0 + br) * DD + bc];
    if (k0 + 32 < 512) {
      __builtin_prefetch(&A[(size_t)ar0 * NN + k0 + 32 + ac0], 0, 3);
      __builtin_prefetch(&Bg[(size_t)(k0 + 32 + br) * DD + bc], 0, 3);
    }
    __syncthreads();
    v16h a = load_a_frag(&As[wave * 16][0], 40);
    for (int j = 0; j < 4; ++j) {
      v16h b = load_b_frag(&Bs[0][j * 16], 72);
      acc[j] = WMMA_F16(a, b, acc[j]);
    }
    __syncthreads();
  }
  int hi = (lane & 16) ? 8 : 0, cl = lane & 15;
  for (int j = 0; j < 4; ++j) {
    for (int r = 0; r < 8; ++r) {
      int m = m0 + wave * 16 + r + hi;
      atomicAdd(&ctx[((size_t)bh * MMF + m) * DHD + j * 16 + cl], acc[j][r]);
    }
  }
}

// ---------------- attn = (qp @ ctx) * d_inv, written [bn, h*64+d] ----------------
__global__ void __launch_bounds__(256)
attn_k(const half_t* __restrict__ qp, const half_t* __restrict__ ctx16,
       const float* __restrict__ dinv, half_t* __restrict__ attn)
{
  __shared__ __align__(16) half_t As[128][40];
  __shared__ __align__(16) half_t Bs[32][72];
  int n0 = blockIdx.x * 128;
  int bh = blockIdx.y, b = bh >> 3, h = bh & 7;
  const half_t* A  = qp + ((size_t)bh * NN + n0) * MMF;
  const half_t* Bg = ctx16 + (size_t)bh * MMF * DHD;
  int t = threadIdx.x;
  int ar0 = t >> 2, ac0 = (t & 3) * 8;
  int ar1 = ar0 + 64;
  int br = t >> 3, bc = (t & 7) * 8;
  int wave = t >> 5, lane = t & 31;
  v8f acc[4] = {{}, {}, {}, {}};
  for (int k0 = 0; k0 < MMF; k0 += 32) {
    *(v8h*)&As[ar0][ac0] = *(const v8h*)&A[(size_t)ar0 * MMF + k0 + ac0];
    *(v8h*)&As[ar1][ac0] = *(const v8h*)&A[(size_t)ar1 * MMF + k0 + ac0];
    *(v8h*)&Bs[br][bc]   = *(const v8h*)&Bg[(size_t)(k0 + br) * DHD + bc];
    if (k0 + 32 < MMF) {
      __builtin_prefetch(&A[(size_t)ar0 * MMF + k0 + 32 + ac0], 0, 3);
      __builtin_prefetch(&Bg[(size_t)(k0 + 32 + br) * DHD + bc], 0, 3);
    }
    __syncthreads();
    v16h a = load_a_frag(&As[wave * 16][0], 40);
    for (int j = 0; j < 4; ++j) {
      v16h b = load_b_frag(&Bs[0][j * 16], 72);
      acc[j] = WMMA_F16(a, b, acc[j]);
    }
    __syncthreads();
  }
  int hi = (lane & 16) ? 8 : 0, cl = lane & 15;
  for (int r = 0; r < 8; ++r) {
    int n = n0 + wave * 16 + r + hi;
    float di = dinv[(size_t)bh * NN + n];
    size_t obase = ((size_t)(b * NN + n)) * DD + h * DHD;
    for (int j = 0; j < 4; ++j)
      attn[obase + j * 16 + cl] = (half_t)(acc[j][r] * di);
  }
}

// ---------------- small helper kernels ----------------
__global__ void ln_k(const float* __restrict__ X, const float* __restrict__ s,
                     const float* __restrict__ bsc, half_t* __restrict__ out)
{
  int row = blockIdx.x, t = threadIdx.x;
  const float* xr = X + (size_t)row * DD;
  float a0 = xr[t], a1 = xr[t + 256];
  __shared__ float red[256];
  red[t] = a0 + a1;
  __syncthreads();
  for (int o = 128; o > 0; o >>= 1) { if (t < o) red[t] += red[t + o]; __syncthreads(); }
  float mean = red[0] * (1.0f / DD);
  __syncthreads();
  float d0 = a0 - mean, d1 = a1 - mean;
  red[t] = d0 * d0 + d1 * d1;
  __syncthreads();
  for (int o = 128; o > 0; o >>= 1) { if (t < o) red[t] += red[t + o]; __syncthreads(); }
  float inv = rsqrtf(red[0] * (1.0f / DD) + 1e-5f);
  out[(size_t)row * DD + t]       = (half_t)(d0 * inv * s[t] + bsc[t]);
  out[(size_t)row * DD + t + 256] = (half_t)(d1 * inv * s[t + 256] + bsc[t + 256]);
}

__global__ void diag_k(const half_t* __restrict__ X, float* __restrict__ out) {
  int idx = blockIdx.x * 8 + (threadIdx.x >> 5);   // (bh,n) flat
  int lane = threadIdx.x & 31;
  int bh = idx >> 13, n = idx & (NN - 1);
  int b = bh >> 3, h = bh & 7;
  const half_t* p = X + ((size_t)(b * NN + n)) * DD + h * DHD + lane * 2;
  float x0 = (float)p[0], x1 = (float)p[1];
  float sum = x0 * x0 + x1 * x1;
  for (int o = 16; o > 0; o >>= 1) sum += __shfl_xor(sum, o);
  if (lane == 0) out[idx] = 0.0625f * sum;   // 0.5 * dn^2, dn^2 = 64^-0.5
}

__global__ void ksum_k(const half_t* __restrict__ kpT, float* __restrict__ ksum) {
  int row = blockIdx.x;  // bh*256 + m
  float sum = 0.f;
  for (int i = threadIdx.x; i < NN; i += 256) sum += (float)kpT[(size_t)row * NN + i];
  __shared__ float red[256];
  red[threadIdx.x] = sum;
  __syncthreads();
  for (int o = 128; o > 0; o >>= 1) { if (threadIdx.x < o) red[threadIdx.x] += red[threadIdx.x + o]; __syncthreads(); }
  if (threadIdx.x == 0) ksum[row] = red[0];
}

__global__ void dinv_k(const half_t* __restrict__ qp, const float* __restrict__ ksum,
                       float* __restrict__ dinv) {
  int idx = blockIdx.x * 8 + (threadIdx.x >> 5);
  int lane = threadIdx.x & 31;
  int bh = idx >> 13;
  float sum = 0.f;
  for (int j = 0; j < 8; ++j) {
    int m = lane + j * 32;
    sum += (float)qp[(size_t)idx * MMF + m] * ksum[bh * MMF + m];
  }
  for (int o = 16; o > 0; o >>= 1) sum += __shfl_xor(sum, o);
  if (lane == 0) dinv[idx] = 1.0f / sum;
}

__global__ void f2h_k(const float* __restrict__ in, half_t* __restrict__ out, int n) {
  int i = blockIdx.x * 256 + threadIdx.x;
  if (i < n) out[i] = (half_t)in[i];
}
__global__ void copy_k(const float* __restrict__ in, float* __restrict__ out, int n) {
  int i = blockIdx.x * 256 + threadIdx.x;
  if (i < n) out[i] = in[i];
}
__global__ void zero_k(float* __restrict__ p, int n) {
  int i = blockIdx.x * 256 + threadIdx.x;
  if (i < n) p[i] = 0.0f;
}
__global__ void projT_k(const float* __restrict__ proj, half_t* __restrict__ out) {
  int i = blockIdx.x * 256 + threadIdx.x;   // i = k*256 + m
  int k = i >> 8, m = i & 255;
  out[i] = (half_t)(proj[(size_t)m * DHD + k] * 0.35355339059f);  // 64^-0.25
}

// ---------------- host pipeline ----------------
extern "C" void kernel_launch(void* const* d_in, const int* in_sizes, int n_in,
                              void* d_out, int out_size, void* d_ws, size_t ws_size,
                              hipStream_t stream)
{
  (void)in_sizes; (void)n_in; (void)out_size; (void)ws_size;
  const float* x    = (const float*)d_in[0];
  const float* ln1s = (const float*)d_in[1];
  const float* ln1b = (const float*)d_in[2];
  const float* Wq   = (const float*)d_in[3];
  const float* bq   = (const float*)d_in[4];
  const float* Wk   = (const float*)d_in[5];
  const float* bk   = (const float*)d_in[6];
  const float* Wv   = (const float*)d_in[7];
  const float* bv   = (const float*)d_in[8];
  const float* Wo   = (const float*)d_in[9];
  const float* bo   = (const float*)d_in[10];
  const float* proj = (const float*)d_in[11];
  const float* ln2s = (const float*)d_in[12];
  const float* ln2b = (const float*)d_in[13];
  const float* W1   = (const float*)d_in[14];
  const float* b1   = (const float*)d_in[15];
  const float* W2   = (const float*)d_in[16];
  const float* b2   = (const float*)d_in[17];
  float* xo = (float*)d_out;

  char* p = (char*)d_ws;
  auto alloc = [&](size_t bytes) -> void* {
    void* r = (void*)p; p += (bytes + 255) & ~(size_t)255; return r;
  };
  const size_t BN = (size_t)BB * NN;
  half_t* h16   = (half_t*)alloc(BN * DD * 2);
  half_t* q16   = (half_t*)alloc(BN * DD * 2);
  half_t* k16   = (half_t*)alloc(BN * DD * 2);
  half_t* v16p  = (half_t*)alloc(BN * DD * 2);
  half_t* at16  = (half_t*)alloc(BN * DD * 2);
  half_t* ff16  = (half_t*)alloc(BN * 4 * DD * 2);
  half_t* qp16  = (half_t*)alloc((size_t)BB * HH * NN * MMF * 2);
  half_t* kpT16 = (half_t*)alloc((size_t)BB * HH * MMF * NN * 2);
  half_t* wq16  = (half_t*)alloc((size_t)DD * DD * 2);
  half_t* wk16  = (half_t*)alloc((size_t)DD * DD * 2);
  half_t* wv16  = (half_t*)alloc((size_t)DD * DD * 2);
  half_t* wo16  = (half_t*)alloc((size_t)DD * DD * 2);
  half_t* w116  = (half_t*)alloc((size_t)DD * 4 * DD * 2);
  half_t* w216  = (half_t*)alloc((size_t)4 * DD * DD * 2);
  half_t* pT16  = (half_t*)alloc((size_t)DHD * MMF * 2);
  half_t* ctx16 = (half_t*)alloc((size_t)BB * HH * MMF * DHD * 2);
  float*  ctxf  = (float*)alloc((size_t)BB * HH * MMF * DHD * 4);
  float*  dgq   = (float*)alloc((size_t)BB * HH * NN * 4);
  float*  dgk   = (float*)alloc((size_t)BB * HH * NN * 4);
  float*  dinv  = (float*)alloc((size_t)BB * HH * NN * 4);
  float*  ksumf = (float*)alloc((size_t)BB * HH * MMF * 4);
  unsigned* stab = (unsigned*)alloc((size_t)BB * HH * 4);

  copy_k<<<(int)(BN * DD / 256), 256, 0, stream>>>(x, xo, (int)(BN * DD));

  dim3 g512(128, 8), gff(128, 32);
  dim3 gfeat(NN / 16, BB * HH);
  for (int l = 0; l < LL; ++l) {
    const float* Wq_l = Wq + (size_t)l * DD * DD;
    const float* Wk_l = Wk + (size_t)l * DD * DD;
    const float* Wv_l = Wv + (size_t)l * DD * DD;
    const float* Wo_l = Wo + (size_t)l * DD * DD;
    const float* W1_l = W1 + (size_t)l * DD * 4 * DD;
    const float* W2_l = W2 + (size_t)l * 4 * DD * DD;
    const float* pj_l = proj + (size_t)l * MMF * DHD;
    const float* bq_l = bq + (size_t)l * DD;
    const float* bk_l = bk + (size_t)l * DD;
    const float* bv_l = bv + (size_t)l * DD;
    const float* bo_l = bo + (size_t)l * DD;
    const float* b1_l = b1 + (size_t)l * 4 * DD;
    const float* b2_l = b2 + (size_t)l * DD;

    f2h_k<<<1024, 256, 0, stream>>>(Wq_l, wq16, DD * DD);
    f2h_k<<<1024, 256, 0, stream>>>(Wk_l, wk16, DD * DD);
    f2h_k<<<1024, 256, 0, stream>>>(Wv_l, wv16, DD * DD);
    f2h_k<<<1024, 256, 0, stream>>>(Wo_l, wo16, DD * DD);
    f2h_k<<<4096, 256, 0, stream>>>(W1_l, w116, DD * 4 * DD);
    f2h_k<<<4096, 256, 0, stream>>>(W2_l, w216, 4 * DD * DD);
    projT_k<<<64, 256, 0, stream>>>(pj_l, pT16);

    // attention block
    ln_k<<<(int)BN, 256, 0, stream>>>(xo, ln1s + (size_t)l * DD, ln1b + (size_t)l * DD, h16);
    gemm16<<<g512, 256, 0, stream>>>(h16, DD, wq16, DD, bq_l, q16, nullptr, DD, DD, 0);
    gemm16<<<g512, 256, 0, stream>>>(h16, DD, wk16, DD, bk_l, k16, nullptr, DD, DD, 0);
    gemm16<<<g512, 256, 0, stream>>>(h16, DD, wv16, DD, bv_l, v16p, nullptr, DD, DD, 0);
    diag_k<<<16384, 256, 0, stream>>>(q16, dgq);
    diag_k<<<16384, 256, 0, stream>>>(k16, dgk);
    zero_k<<<1, 64, 0, stream>>>((float*)stab, BB * HH);
    feat_k<<<gfeat, 256, 0, stream>>>(k16, pT16, nullptr, stab, nullptr, 0);
    feat_k<<<gfeat, 256, 0, stream>>>(q16, pT16, dgq, nullptr, qp16, 1);
    feat_k<<<gfeat, 256, 0, stream>>>(k16, pT16, dgk, stab, kpT16, 2);
    ksum_k<<<BB * HH * MMF, 256, 0, stream>>>(kpT16, ksumf);
    dinv_k<<<16384, 256, 0, stream>>>(qp16, ksumf, dinv);
    zero_k<<<1024, 256, 0, stream>>>(ctxf, BB * HH * MMF * DHD);
    ctx_k<<<dim3(2, 16, BB * HH), 256, 0, stream>>>(kpT16, v16p, ctxf);
    f2h_k<<<1024, 256, 0, stream>>>(ctxf, ctx16, BB * HH * MMF * DHD);
    attn_k<<<dim3(NN / 128, BB * HH), 256, 0, stream>>>(qp16, ctx16, dinv, at16);
    gemm16<<<g512, 256, 0, stream>>>(at16, DD, wo16, DD, bo_l, nullptr, xo, DD, DD, 0);

    // feedforward block
    ln_k<<<(int)BN, 256, 0, stream>>>(xo, ln2s + (size_t)l * DD, ln2b + (size_t)l * DD, h16);
    gemm16<<<gff, 256, 0, stream>>>(h16, DD, w116, 4 * DD, b1_l, ff16, nullptr, 4 * DD, DD, 1);
    gemm16<<<g512, 256, 0, stream>>>(ff16, 4 * DD, w216, DD, b2_l, nullptr, xo, DD, 4 * DD, 0);
  }
}
